// MultiHeadAttention_14542759264693
// MI455X (gfx1250) — compile-verified, matrix operand order checked by reading, not run on hardware
//
#include <hip/hip_runtime.h>

typedef __attribute__((ext_vector_type(16))) __bf16 v16bf;
typedef __attribute__((ext_vector_type(8)))  __bf16 v8bf;
typedef __attribute__((ext_vector_type(4)))  __bf16 v4bf;
typedef __attribute__((ext_vector_type(8)))  float  v8f;

#define DM   512
#define SEQ  2048
#define NH   8
#define HD   64
#define LSTR 40   // padded LDS row stride (elements); 80 bytes = multiple of 16

// Load a 16-element bf16 A/B fragment for v_wmma_f32_16x16x32_bf16.
// Lane holds row/col (lane%16); g = lane/16 selects k-halves:
//   elems 0..7  -> k = g*8 .. g*8+7      (contiguous 16B)
//   elems 8..15 -> k = 16+g*8 .. 23+g*8  (contiguous 16B)
__device__ __forceinline__ v16bf frag_ld(const __bf16* p, int g) {
  v8bf lo = *(const v8bf*)(p + g * 8);
  v8bf hi = *(const v8bf*)(p + g * 8 + 16);
  v16bf r;
#pragma unroll
  for (int i = 0; i < 8; ++i) { r[i] = lo[i]; r[i + 8] = hi[i]; }
  return r;
}

// ---- DPP16 row rotate (VALU, no DS unit) for 16-lane all-reduce -----------
// ctrl 0x120+n = ROW_ROR:n ; rows of 16 lanes, halves stay independent.
template <int CTRL>
__device__ __forceinline__ float dpp_rot(float x) {
  return __builtin_bit_cast(
      float, __builtin_amdgcn_update_dpp(0, __builtin_bit_cast(int, x), CTRL,
                                         0xf, 0xf, true));
}
__device__ __forceinline__ float rowmax16(float x) {
  x = fmaxf(x, dpp_rot<0x121>(x));
  x = fmaxf(x, dpp_rot<0x122>(x));
  x = fmaxf(x, dpp_rot<0x124>(x));
  x = fmaxf(x, dpp_rot<0x128>(x));
  return x;
}
__device__ __forceinline__ float rowsum16(float x) {
  x += dpp_rot<0x121>(x);
  x += dpp_rot<0x122>(x);
  x += dpp_rot<0x124>(x);
  x += dpp_rot<0x128>(x);
  return x;
}

// ---------------------------------------------------------------------------
// GEMM: Y[M,512] = X[M,512] @ W[512,512] + bias, M = 8192.
// Block = 64x64 tile, 8 waves, each wave a 16x32 sub-tile (2 accumulators).
// MODE 0: bf16 out in [B,H,S,Dh]       (Q, K)
// MODE 1: bf16 out in [B,H,Dh,S]       (V transposed)
// MODE 2: f32  out row-major [M,512]   (final projection)
// ---------------------------------------------------------------------------
template <int MODE, typename InT>
__global__ __launch_bounds__(256) void gemm512(const InT* __restrict__ X,
                                               const float* __restrict__ W,
                                               const float* __restrict__ bias,
                                               void* __restrict__ outp) {
  __shared__ __bf16 Al[64 * LSTR];
  __shared__ __bf16 Bl[64 * LSTR];
  const int t = threadIdx.x;
  const int lane = t & 31, wave = t >> 5;
  const int wr = wave >> 1, wc = wave & 1;   // 4x2 wave grid
  const int r = lane & 15, g = lane >> 4;
  const int m0 = blockIdx.y * 64, n0 = blockIdx.x * 64;

  const v8f vzero = {0.f, 0.f, 0.f, 0.f, 0.f, 0.f, 0.f, 0.f};
  v8f acc[2];
  acc[0] = vzero; acc[1] = vzero;

  for (int kc = 0; kc < DM; kc += 32) {
    __syncthreads();
    // stage A tile 64x32 (convert fp32 -> bf16 if needed)
#pragma unroll
    for (int p = 0; p < 2; ++p) {
      const int lin = t + p * 256;          // float4 index over 64x32
      const int row = lin >> 3;
      const int col = (lin & 7) << 2;
      __bf16* dst = &Al[row * LSTR + col];
      if constexpr (__is_same(InT, float)) {
        const float4 s4 = *(const float4*)(X + (size_t)(m0 + row) * DM + kc + col);
        dst[0] = (__bf16)s4.x; dst[1] = (__bf16)s4.y;
        dst[2] = (__bf16)s4.z; dst[3] = (__bf16)s4.w;
      } else {
        *(v4bf*)dst = *(const v4bf*)(X + (size_t)(m0 + row) * DM + kc + col);
      }
    }
    // stage B tile transposed: Bl[n][k] = W[kc+k][n0+n]
#pragma unroll
    for (int p = 0; p < 2; ++p) {
      const int lin = t + p * 256;          // float4 index over 32x64
      const int k = lin >> 4;
      const int n = (lin & 15) << 2;
      const float4 w4 = *(const float4*)(W + (size_t)(kc + k) * DM + n0 + n);
      Bl[(n + 0) * LSTR + k] = (__bf16)w4.x;
      Bl[(n + 1) * LSTR + k] = (__bf16)w4.y;
      Bl[(n + 2) * LSTR + k] = (__bf16)w4.z;
      Bl[(n + 3) * LSTR + k] = (__bf16)w4.w;
    }
    __syncthreads();

    const v16bf af = frag_ld(&Al[(wr * 16 + r) * LSTR], g);
#pragma unroll
    for (int tt = 0; tt < 2; ++tt) {
      const v16bf bf_ = frag_ld(&Bl[(wc * 32 + tt * 16 + r) * LSTR], g);
      acc[tt] = __builtin_amdgcn_wmma_f32_16x16x32_bf16(
          false, af, false, bf_, (short)0, acc[tt], false, false);
    }
  }

  // epilogue: C/D layout -> element (m = v + 8*g, n = lane%16)
#pragma unroll
  for (int tt = 0; tt < 2; ++tt) {
#pragma unroll
    for (int v = 0; v < 8; ++v) {
      const int row = m0 + wr * 16 + v + 8 * g;
      const int col = n0 + wc * 32 + tt * 16 + r;
      const float val = acc[tt][v] + bias[col];
      if constexpr (MODE == 2) {
        ((float*)outp)[(size_t)row * DM + col] = val;
      } else {
        const int b = row >> 11, s = row & (SEQ - 1);
        const int h = col >> 6, d = col & (HD - 1);
        __bf16* ob = (__bf16*)outp;
        if constexpr (MODE == 0)
          ob[((size_t)(b * NH + h) * SEQ + s) * HD + d] = (__bf16)val;
        else
          ob[((size_t)(b * NH + h) * HD + d) * SEQ + s] = (__bf16)val;
      }
    }
  }
}

// ---------------------------------------------------------------------------
// Flash attention: one wave = one 16-query tile of one (b,h).
// Q,K in [B,H,S,Dh] bf16 ; V transposed [B,H,Dh,S] bf16 ; mask [B,1,S,S] i32.
// K/V are L2-resident (8 MB each) so per-wave re-reads are cheap.
// Softmax is carried in the log2 domain: v_exp_f32 is base-2 natively.
// ---------------------------------------------------------------------------
__global__ __launch_bounds__(256) void flash_attn(const __bf16* __restrict__ Qb,
                                                  const __bf16* __restrict__ Kb,
                                                  const __bf16* __restrict__ Vt,
                                                  const int* __restrict__ mask,
                                                  __bf16* __restrict__ Ao) {
  __shared__ __bf16 Plds[8][16 * 32];   // per-wave P staging (C-layout -> A-frag)
  const int t = threadIdx.x, lane = t & 31, wave = t >> 5;
  const int r = lane & 15, g = lane >> 4;
  const int gw = blockIdx.x * 8 + wave;      // 0..4095
  const int bh = gw >> 7, qt = gw & 127;
  const int b = bh >> 3, h = bh & 7;
  const int q0 = qt * 16;

  // scale = 1/sqrt(Dh) folded together with log2(e): softmax in log2 domain
  const float SC = 0.125f * 1.4426950408889634f;

  const __bf16* Qp = Qb + ((size_t)bh * SEQ + q0) * HD;
  const __bf16* Kp = Kb + (size_t)bh * SEQ * HD;
  const __bf16* Vp = Vt + (size_t)bh * HD * SEQ;
  const int*    Mp = mask + ((size_t)b * SEQ + q0) * SEQ;

  v16bf qf[2];
#pragma unroll
  for (int c = 0; c < 2; ++c) qf[c] = frag_ld(Qp + r * HD + c * 32, g);

  const v8f vzero = {0.f, 0.f, 0.f, 0.f, 0.f, 0.f, 0.f, 0.f};
  v8f o[4];
  float m_i[8], l_i[8];
#pragma unroll
  for (int j = 0; j < 4; ++j) o[j] = vzero;
#pragma unroll
  for (int v = 0; v < 8; ++v) { m_i[v] = -1e30f; l_i[v] = 0.f; }

  __bf16* pw = &Plds[wave][0];

  for (int kc = 0; kc < SEQ; kc += 32) {
    // ---- S = Q @ K^T for 16q x 32k (4 WMMAs) ----
    v8f s[2];
    s[0] = vzero; s[1] = vzero;
#pragma unroll
    for (int tt = 0; tt < 2; ++tt) {
      const __bf16* kb = Kp + (size_t)(kc + tt * 16 + r) * HD;
#pragma unroll
      for (int c = 0; c < 2; ++c) {
        const v16bf kf = frag_ld(kb + c * 32, g);
        s[tt] = __builtin_amdgcn_wmma_f32_16x16x32_bf16(
            false, qf[c], false, kf, (short)0, s[tt], false, false);
      }
    }
    if (kc + 32 < SEQ)  // prefetch next chunk's K row for this lane
      __builtin_prefetch(Kp + (size_t)(kc + 32 + r) * HD, 0, 1);

    // ---- mask, scale, online softmax (DPP row reductions, no DS ops) ----
#pragma unroll
    for (int v = 0; v < 8; ++v) {
      const int mrow = v + 8 * g;
      const int* mp = Mp + (size_t)mrow * SEQ + kc + r;
      float s0 = s[0][v] * SC;
      float s1 = s[1][v] * SC;
      s0 = (mp[0]  == 0) ? -1e9f : s0;
      s1 = (mp[16] == 0) ? -1e9f : s1;
      const float rm = rowmax16(fmaxf(s0, s1));
      const float mn = fmaxf(m_i[v], rm);
      const float f  = __builtin_amdgcn_exp2f(m_i[v] - mn);
      const float p0 = __builtin_amdgcn_exp2f(s0 - mn);
      const float p1 = __builtin_amdgcn_exp2f(s1 - mn);
      const float rs = rowsum16(p0 + p1);
      l_i[v] = l_i[v] * f + rs;
      m_i[v] = mn;
#pragma unroll
      for (int j = 0; j < 4; ++j) o[j][v] *= f;
      pw[mrow * 32 + r]      = (__bf16)p0;   // stage P in [m][k_local]
      pw[mrow * 32 + 16 + r] = (__bf16)p1;
    }

    // ---- O += P @ V for 16q x 64d (4 WMMAs); same-wave DS ops are in-order ----
    const v16bf pf = frag_ld(&pw[r * 32], g);
#pragma unroll
    for (int j = 0; j < 4; ++j) {
      const v16bf vf = frag_ld(Vp + (size_t)(j * 16 + r) * SEQ + kc, g);
      o[j] = __builtin_amdgcn_wmma_f32_16x16x32_bf16(
          false, pf, false, vf, (short)0, o[j], false, false);
    }
  }

  // ---- normalize and write attn output in [B*S, 512] bf16 ----
#pragma unroll
  for (int v = 0; v < 8; ++v) {
    const float inv = 1.0f / l_i[v];
    const int row = b * SEQ + q0 + v + 8 * g;
#pragma unroll
    for (int j = 0; j < 4; ++j)
      Ao[(size_t)row * DM + h * HD + j * 16 + r] = (__bf16)(o[j][v] * inv);
  }
}

// ---------------------------------------------------------------------------
extern "C" void kernel_launch(void* const* d_in, const int* in_sizes, int n_in,
                              void* d_out, int out_size, void* d_ws, size_t ws_size,
                              hipStream_t stream) {
  (void)in_sizes; (void)n_in; (void)out_size; (void)ws_size;
  const float* q  = (const float*)d_in[0];
  const float* k  = (const float*)d_in[1];
  const float* v  = (const float*)d_in[2];
  const int*   mk = (const int*)d_in[3];
  const float* Wq = (const float*)d_in[4];
  const float* bq = (const float*)d_in[5];
  const float* Wk = (const float*)d_in[6];
  const float* bk = (const float*)d_in[7];
  const float* Wv = (const float*)d_in[8];
  const float* bv = (const float*)d_in[9];
  const float* Wo = (const float*)d_in[10];
  const float* bo = (const float*)d_in[11];

  const size_t E = (size_t)4 * NH * SEQ * HD;  // 4,194,304 bf16 elements
  __bf16* Qb = (__bf16*)d_ws;
  __bf16* Kb = Qb + E;
  __bf16* Vt = Kb + E;
  __bf16* Ao = Vt + E;   // total 32 MB of workspace

  dim3 gg(DM / 64, (4 * SEQ) / 64), bb(256);
  gemm512<0, float><<<gg, bb, 0, stream>>>(q, Wq, bq, (void*)Qb);
  gemm512<0, float><<<gg, bb, 0, stream>>>(k, Wk, bk, (void*)Kb);
  gemm512<1, float><<<gg, bb, 0, stream>>>(v, Wv, bv, (void*)Vt);
  flash_attn<<<(4 * NH * (SEQ / 16)) / 8, 256, 0, stream>>>(Qb, Kb, Vt, mk, Ao);
  gemm512<2, __bf16><<<gg, bb, 0, stream>>>(Ao, Wo, bo, d_out);
}
